// ScoreBlock_27685359190683
// MI455X (gfx1250) — matrix operand order repaired and checked by reading, not compiled, not run on hardware
//
#include <hip/hip_runtime.h>

typedef __bf16 bf16_t;
typedef __attribute__((ext_vector_type(16))) __bf16 v16bf;
typedef __attribute__((ext_vector_type(8)))  __bf16 v8bf;
typedef __attribute__((ext_vector_type(4)))  __bf16 v4bf;
typedef __attribute__((ext_vector_type(8)))  float  v8f;

#define B_     8
#define NB_    2048
#define NS_    1024
#define C_     768
#define HID_   3072
#define NP_    1024
#define NH_    6
#define HD_    128
#define KK_    128
#define NHALF_ 512

// ---------------- fragment loaders (CDNA5 16-bit WMMA layouts) ----------------
// A 16x32 (MxK): lane m=lane&15, half=lane>>4 holds K = {half*8+0..7, 16+half*8+0..7}
static __device__ inline v16bf load_frag_a(const bf16_t* base, int stride, int lane) {
  const int m = lane & 15, h = lane >> 4;
  v8bf lo = *(const v8bf*)(base + (size_t)m * stride + h * 8);
  v8bf hi = *(const v8bf*)(base + (size_t)m * stride + 16 + h * 8);
  return __builtin_shufflevector(lo, hi, 0,1,2,3,4,5,6,7,8,9,10,11,12,13,14,15);
}
// B 32x16 (KxN): lane n=lane&15, half=lane>>4 holds K = {half*8+0..7, 16+half*8+0..7}
static __device__ inline v16bf load_frag_b(const bf16_t* base, int stride, int lane) {
  const int n = lane & 15, h = lane >> 4;
  v8bf lo = *(const v8bf*)(base + (size_t)n * stride + h * 8);
  v8bf hi = *(const v8bf*)(base + (size_t)n * stride + 16 + h * 8);
  return __builtin_shufflevector(lo, hi, 0,1,2,3,4,5,6,7,8,9,10,11,12,13,14,15);
}
static __device__ inline v8f wmma_bf16(v16bf a, v16bf b, v8f c) {
  return __builtin_amdgcn_wmma_f32_16x16x32_bf16(false, a, false, b, (short)0, c, false, false);
}

// ---------------- block reductions (wave32, 8 waves / 256 threads) ----------------
static __device__ inline float block_sum(float v, float* sbuf) {
  const int tid = threadIdx.x;
  for (int off = 16; off > 0; off >>= 1) v += __shfl_xor(v, off, 32);
  if ((tid & 31) == 0) sbuf[tid >> 5] = v;
  __syncthreads();
  float t = 0.f;
  for (int i = 0; i < 8; ++i) t += sbuf[i];
  __syncthreads();
  return t;
}
static __device__ inline float block_max(float v, float* sbuf) {
  const int tid = threadIdx.x;
  for (int off = 16; off > 0; off >>= 1) v = fmaxf(v, __shfl_xor(v, off, 32));
  if ((tid & 31) == 0) sbuf[tid >> 5] = v;
  __syncthreads();
  float t = -1e30f;
  for (int i = 0; i < 8; ++i) t = fmaxf(t, sbuf[i]);
  __syncthreads();
  return t;
}

// ---------------- elementwise utilities ----------------
__global__ void f32_to_bf16_kernel(const float* __restrict__ in, bf16_t* __restrict__ out, long n) {
  const long i = (long)blockIdx.x * blockDim.x + threadIdx.x;
  if (i < n) out[i] = (bf16_t)in[i];
}
__global__ void copy_f32_kernel(const float* __restrict__ in, float* __restrict__ out, long n) {
  const long i = (long)blockIdx.x * blockDim.x + threadIdx.x;
  if (i < n) out[i] = in[i];
}
__global__ void zero_f32_kernel(float* __restrict__ p, long n) {
  const long i = (long)blockIdx.x * blockDim.x + threadIdx.x;
  if (i < n) p[i] = 0.f;
}

// ---------------- LayerNorm + cast to bf16 ----------------
__global__ __launch_bounds__(256) void ln_cast_kernel(
    const float* __restrict__ x, const float* __restrict__ w, const float* __restrict__ bb,
    bf16_t* __restrict__ out, const int C) {
  __shared__ float sbuf[8];
  const int row = blockIdx.x, tid = threadIdx.x;
  const float* xr = x + (size_t)row * C;
  float s = 0.f, ss = 0.f;
  for (int c = tid; c < C; c += 256) { const float v = xr[c]; s += v; ss += v * v; }
  s  = block_sum(s,  sbuf);
  ss = block_sum(ss, sbuf);
  const float mu   = s / C;
  const float var  = ss / C - mu * mu;
  const float rstd = rsqrtf(var + 1e-5f);
  bf16_t* orow = out + (size_t)row * C;
  for (int c = tid; c < C; c += 256)
    orow[c] = (bf16_t)((xr[c] - mu) * rstd * w[c] + bb[c]);
}

// ---------------- generic bf16 WMMA GEMM, 128x128 tile ----------------
// Double-buffered B tile in LDS; next tile's global loads issued before the
// current tile's WMMAs so HBM latency hides behind matrix math; B fragments
// are register-batched so the 8 WMMAs issue back-to-back after one dscnt wait.
// mode 0: outbf = acc(+bias)            mode 1: outbf = gelu(acc+bias)
// mode 2: outf  += (tt[row]==ttv) ? acc+bias : 0     mode 3: outf += acc+bias
__global__ __launch_bounds__(256) void gemm_bf16_kernel(
    const bf16_t* __restrict__ A, const bf16_t* __restrict__ Bw,
    const int M, const int N, const int K,
    const float* __restrict__ bias, const int mode,
    const int* __restrict__ tt, const int ttv,
    float* __restrict__ outf, bf16_t* __restrict__ outbf) {
  const int tid = threadIdx.x, wave = tid >> 5, lane = tid & 31;
  const int lm = lane & 15, lh = lane >> 4;
  const int n0 = blockIdx.x * 128, m0 = blockIdx.y * 128;
  __shared__ alignas(16) bf16_t Bt[2][128 * 40];   // [n][k], k contiguous, stride 40

  v8f acc[8];
  #pragma unroll
  for (int nt = 0; nt < 8; ++nt)
    #pragma unroll
    for (int i = 0; i < 8; ++i) acc[nt][i] = 0.f;

  const bf16_t* arow = A + (size_t)(m0 + wave * 16) * K;

  // per-thread staging coordinates: 4 chunks of 4 bf16 (32x128 tile / 256 thr)
  int skk[4], sn4[4];
  #pragma unroll
  for (int it = 0; it < 4; ++it) {
    const int c = tid + it * 256;
    skk[it] = c >> 5;          // 0..31
    sn4[it] = (c & 31) << 2;   // 0,4,..,124
  }

  v4bf st[4];
  // prologue: stage tile 0
  #pragma unroll
  for (int it = 0; it < 4; ++it)
    st[it] = *(const v4bf*)(Bw + (size_t)skk[it] * N + n0 + sn4[it]);
  #pragma unroll
  for (int it = 0; it < 4; ++it) {
    #pragma unroll
    for (int j = 0; j < 4; ++j) Bt[0][(sn4[it] + j) * 40 + skk[it]] = st[it][j];
  }
  __syncthreads();

  int cur = 0;
  for (int k0 = 0; k0 < K; k0 += 32) {
    const bool more = (k0 + 32) < K;
    // issue next tile's global loads early
    if (more) {
      #pragma unroll
      for (int it = 0; it < 4; ++it)
        st[it] = *(const v4bf*)(Bw + (size_t)(k0 + 32 + skk[it]) * N + n0 + sn4[it]);
    }
    // register-batch all fragments, then back-to-back WMMAs
    const v16bf af = load_frag_a(arow + k0, K, lane);
    v16bf bfv[8];
    #pragma unroll
    for (int nt = 0; nt < 8; ++nt)
      bfv[nt] = load_frag_b(&Bt[cur][nt * 16 * 40], 40, lane);
    #pragma unroll
    for (int nt = 0; nt < 8; ++nt)
      acc[nt] = wmma_bf16(af, bfv[nt], acc[nt]);
    // commit next tile to the other buffer
    if (more) {
      #pragma unroll
      for (int it = 0; it < 4; ++it) {
        #pragma unroll
        for (int j = 0; j < 4; ++j) Bt[cur ^ 1][(sn4[it] + j) * 40 + skk[it]] = st[it][j];
      }
    }
    __syncthreads();
    cur ^= 1;
  }

  #pragma unroll
  for (int nt = 0; nt < 8; ++nt) {
    #pragma unroll
    for (int i = 0; i < 8; ++i) {
      const int row = m0 + wave * 16 + lh * 8 + i;
      const int col = n0 + nt * 16 + lm;
      float v = acc[nt][i];
      if (bias) v += bias[col];
      if (mode == 0) {
        outbf[(size_t)row * N + col] = (bf16_t)v;
      } else if (mode == 1) {
        v = 0.5f * v * (1.f + erff(v * 0.7071067811865476f));
        outbf[(size_t)row * N + col] = (bf16_t)v;
      } else if (mode == 2) {
        if (tt[row] == ttv) outf[(size_t)row * N + col] += v;
      } else {
        outf[(size_t)row * N + col] += v;
      }
    }
  }
}

// ---------------- flash attention: q(128 rows/block) over all 2048 keys ----------------
__global__ __launch_bounds__(256) void attention_kernel(
    const bf16_t* __restrict__ qg, const bf16_t* __restrict__ kvg, bf16_t* __restrict__ ctx) {
  const int tid = threadIdx.x, wave = tid >> 5, lane = tid & 31;
  const int lm = lane & 15, lh = lane >> 4;
  const int qt = blockIdx.x, h = blockIdx.y, b = blockIdx.z;
  const int q0 = qt * 128;

  __shared__ alignas(16) bf16_t Vt[HD_ * 72];     // [hd][kpos] (64-key tile, stride 72)
  __shared__ alignas(16) bf16_t Ps[8 * 16 * 72];  // per-wave P tile 16x64 (stride 72)

  // preload Q fragments for this wave's 16 rows (contraction over HD=128 in 4 steps)
  v16bf qf[4];
  {
    const bf16_t* qbase = qg + (size_t)(b * NS_ + q0 + wave * 16) * C_ + h * HD_;
    #pragma unroll
    for (int ks = 0; ks < 4; ++ks) qf[ks] = load_frag_a(qbase + ks * 32, C_, lane);
  }
  float mrow[8], lrow[8];
  v8f ob[8];
  #pragma unroll
  for (int i = 0; i < 8; ++i) { mrow[i] = -1e30f; lrow[i] = 0.f; }
  #pragma unroll
  for (int nt = 0; nt < 8; ++nt)
    #pragma unroll
    for (int i = 0; i < 8; ++i) ob[nt][i] = 0.f;

  const float scale = 0.08838834764831845f;  // HD^-0.5
  const bf16_t* kvb = kvg + (size_t)b * NB_ * (2 * C_);

  for (int kt = 0; kt < NB_ / 64; ++kt) {
    // stage V tile (64 keys x 128 hd) transposed into LDS
    for (int c = tid; c < 64 * 16; c += 256) {
      const int m = c >> 4;           // key within tile
      const int kk = (c & 15) << 3;   // hd
      const v8bf vv = *(const v8bf*)(kvb + (size_t)(kt * 64 + m) * (2 * C_) + C_ + h * HD_ + kk);
      #pragma unroll
      for (int j = 0; j < 8; ++j) Vt[(kk + j) * 72 + m] = vv[j];
    }
    __syncthreads();

    // S = Q K^T : 16 rows x 64 keys per wave (K fragments straight from global,
    // batched per HD step so the 4 WMMAs issue together)
    v8f s[4];
    #pragma unroll
    for (int nt = 0; nt < 4; ++nt)
      #pragma unroll
      for (int i = 0; i < 8; ++i) s[nt][i] = 0.f;

    const bf16_t* kbase = kvb + (size_t)(kt * 64) * (2 * C_) + h * HD_;
    #pragma unroll
    for (int ks = 0; ks < 4; ++ks) {
      v16bf kf[4];
      #pragma unroll
      for (int nt = 0; nt < 4; ++nt)
        kf[nt] = load_frag_b(kbase + (size_t)(nt * 16) * (2 * C_) + ks * 32, 2 * C_, lane);
      #pragma unroll
      for (int nt = 0; nt < 4; ++nt)
        s[nt] = wmma_bf16(qf[ks], kf[nt], s[nt]);
    }
    // online softmax per row (row = lh*8+i, reduce over 16-lane half)
    #pragma unroll
    for (int i = 0; i < 8; ++i) {
      float mloc = -1e30f;
      #pragma unroll
      for (int nt = 0; nt < 4; ++nt) { s[nt][i] *= scale; mloc = fmaxf(mloc, s[nt][i]); }
      #pragma unroll
      for (int off = 1; off < 16; off <<= 1) mloc = fmaxf(mloc, __shfl_xor(mloc, off, 32));
      const float mnew  = fmaxf(mrow[i], mloc);
      const float alpha = __expf(mrow[i] - mnew);
      lrow[i] *= alpha;
      #pragma unroll
      for (int nt = 0; nt < 8; ++nt) ob[nt][i] *= alpha;
      float rs = 0.f;
      #pragma unroll
      for (int nt = 0; nt < 4; ++nt) { const float p = __expf(s[nt][i] - mnew); s[nt][i] = p; rs += p; }
      #pragma unroll
      for (int off = 1; off < 16; off <<= 1) rs += __shfl_xor(rs, off, 32);
      lrow[i] += rs;
      mrow[i] = mnew;
    }
    // P -> bf16 in per-wave LDS, then O += P @ V (V fragments batched per kc)
    bf16_t* pw = Ps + wave * 16 * 72;
    #pragma unroll
    for (int nt = 0; nt < 4; ++nt)
      #pragma unroll
      for (int i = 0; i < 8; ++i)
        pw[(lh * 8 + i) * 72 + nt * 16 + lm] = (bf16_t)s[nt][i];
    asm volatile("s_wait_dscnt 0" ::: "memory");
    #pragma unroll
    for (int kc = 0; kc < 2; ++kc) {
      const v16bf af = load_frag_a(pw + kc * 32, 72, lane);
      v16bf vf[8];
      #pragma unroll
      for (int nt = 0; nt < 8; ++nt)
        vf[nt] = load_frag_b(Vt + nt * 16 * 72 + kc * 32, 72, lane);
      #pragma unroll
      for (int nt = 0; nt < 8; ++nt)
        ob[nt] = wmma_bf16(af, vf[nt], ob[nt]);
    }
    __syncthreads();
  }
  #pragma unroll
  for (int nt = 0; nt < 8; ++nt)
    #pragma unroll
    for (int i = 0; i < 8; ++i) {
      const int row = q0 + wave * 16 + lh * 8 + i;
      const int col = h * HD_ + nt * 16 + lm;
      ctx[(size_t)(b * NS_ + row) * C_ + col] = (bf16_t)(ob[nt][i] / lrow[i]);
    }
}

// ---------------- scoring head (small, VALU f32) ----------------
__global__ __launch_bounds__(256) void rownorm_inv_kernel(
    const float* __restrict__ x, float* __restrict__ inv, const int C) {
  __shared__ float sbuf[8];
  const int row = blockIdx.x, tid = threadIdx.x;
  const float* xr = x + (size_t)row * C;
  float ss = 0.f;
  for (int c = tid; c < C; c += 256) { const float v = xr[c]; ss += v * v; }
  ss = block_sum(ss, sbuf);
  if (tid == 0) inv[row] = 1.f / fmaxf(sqrtf(ss), 1e-12f);
}

__global__ __launch_bounds__(256) void gather_tokens_kernel(
    const float* __restrict__ xb, const int* __restrict__ bidx, float* __restrict__ tokens) {
  const int blk = blockIdx.x;
  const int b = blk / NHALF_, i = blk % NHALF_;
  const int id = bidx[b * NP_ + i];
  const bool mk = id >= 0;
  const int idc = mk ? id : 0;
  const float* src = xb + ((size_t)b * NB_ + idc) * C_;
  float* dst = tokens + ((size_t)b * NHALF_ + i) * C_;
  for (int c = threadIdx.x; c < C_; c += 256) dst[c] = mk ? src[c] : 0.f;
}

__global__ __launch_bounds__(256) void token_attn_kernel(
    const float* __restrict__ tokens, const int* __restrict__ bidx, float* __restrict__ agg) {
  __shared__ float ti[C_];
  __shared__ float aw[NHALF_];
  __shared__ float sbuf[8];
  const int blk = blockIdx.x;
  const int b = blk / NHALF_, i = blk % NHALF_;
  const int tid = threadIdx.x;
  const float* trow = tokens + ((size_t)b * NHALF_ + i) * C_;
  for (int c = tid; c < C_; c += 256) ti[c] = trow[c];
  __syncthreads();
  const float sc_scale = 0.036084391824351615f;  // 1/sqrt(768)
  for (int j = tid; j < NHALF_; j += 256) {
    const float* tj = tokens + ((size_t)b * NHALF_ + j) * C_;
    float d = 0.f;
    for (int c = 0; c < C_; ++c) d += ti[c] * tj[c];
    const float kb = (bidx[b * NP_ + j] >= 0) ? 0.f : -1e9f;
    aw[j] = d * sc_scale + kb;
  }
  __syncthreads();
  float mx = -1e30f;
  for (int j = tid; j < NHALF_; j += 256) mx = fmaxf(mx, aw[j]);
  mx = block_max(mx, sbuf);
  float sm = 0.f;
  for (int j = tid; j < NHALF_; j += 256) { const float e = __expf(aw[j] - mx); aw[j] = e; sm += e; }
  sm = block_sum(sm, sbuf);
  const float isum = 1.f / sm;
  float* arow = agg + ((size_t)b * NHALF_ + i) * C_;
  for (int c = tid; c < C_; c += 256) {
    float acc = 0.f;
    for (int j = 0; j < NHALF_; ++j)
      acc += aw[j] * tokens[((size_t)b * NHALF_ + j) * C_ + c];
    arow[c] = acc * isum;
  }
}

__global__ __launch_bounds__(256) void kernel_reduce_kernel(
    const float* __restrict__ agg, const int* __restrict__ bidx,
    float* __restrict__ kout, float* __restrict__ kws, float* __restrict__ hasws) {
  __shared__ float sbuf[8];
  __shared__ float kn[C_];
  const int b = blockIdx.x, tid = threadIdx.x;
  float c = 0.f;
  for (int i = tid; i < NHALF_; i += 256) c += (bidx[b * NP_ + i] >= 0) ? 1.f : 0.f;
  c = block_sum(c, sbuf);
  const int cnt = (int)(c + 0.5f);
  const float denom = (float)(cnt > 0 ? cnt : 1);
  for (int col = tid; col < C_; col += 256) {
    float s = 0.f;
    for (int i = 0; i < NHALF_; ++i)
      if (bidx[b * NP_ + i] >= 0)
        s += agg[((size_t)b * NHALF_ + i) * C_ + col];
    kn[col] = s / denom;
  }
  __syncthreads();
  float ss = 0.f;
  for (int col = tid; col < C_; col += 256) ss += kn[col] * kn[col];
  ss = block_sum(ss, sbuf);
  const float inv = 1.f / fmaxf(sqrtf(ss), 1e-12f);
  const float has = (cnt > 0) ? 1.f : 0.f;
  for (int col = tid; col < C_; col += 256) {
    const float v = has * kn[col] * inv;
    kout[(size_t)b * C_ + col] = v;
    kws[(size_t)b * C_ + col] = v;
  }
  if (tid == 0) hasws[b] = has;
}

__global__ __launch_bounds__(256) void pos_kernel(
    const float* __restrict__ x, const float* __restrict__ inv,
    const float* __restrict__ kws, const float* __restrict__ hasws,
    float* __restrict__ pos) {
  __shared__ float sbuf[8];
  const int blk = blockIdx.x;
  const int b = blk / NS_, n = blk % NS_;
  const int tid = threadIdx.x;
  const float* xr = x + ((size_t)b * NS_ + n) * C_;
  const float* kr = kws + (size_t)b * C_;
  float d = 0.f;
  for (int c = tid; c < C_; c += 256) d += xr[c] * kr[c];
  d = block_sum(d, sbuf);
  if (tid == 0) {
    const float has = hasws[b];
    pos[(size_t)b * NS_ + n] = (has > 0.f) ? (d * inv[b * NS_ + n] + 1.f) * 0.5f : 0.f;
  }
}

__global__ __launch_bounds__(256) void topk_kernel(
    const float* __restrict__ pos, float* __restrict__ sel, int* __restrict__ topi) {
  __shared__ float vals[NS_];
  __shared__ float rv[256];
  __shared__ int   ri[256];
  __shared__ float topv[KK_];
  __shared__ int   topidx[KK_];
  __shared__ int   cnt65;
  const int b = blockIdx.x, tid = threadIdx.x;
  if (tid == 0) cnt65 = 0;
  __syncthreads();
  int lc = 0;
  for (int i = tid; i < NS_; i += 256) {
    const float v = pos[(size_t)b * NS_ + i];
    vals[i] = v;
    if (v > 0.65f) ++lc;
  }
  atomicAdd(&cnt65, lc);
  __syncthreads();
  for (int it = 0; it < KK_; ++it) {
    float bv = -1e30f; int bi = NS_;
    for (int k = 0; k < NS_ / 256; ++k) {
      const int idx = tid + k * 256;
      const float v = vals[idx];
      if (v > bv || (v == bv && idx < bi)) { bv = v; bi = idx; }
    }
    rv[tid] = bv; ri[tid] = bi;
    __syncthreads();
    for (int s2 = 128; s2 > 0; s2 >>= 1) {
      if (tid < s2) {
        const float ov = rv[tid + s2]; const int oi = ri[tid + s2];
        if (ov > rv[tid] || (ov == rv[tid] && oi < ri[tid])) { rv[tid] = ov; ri[tid] = oi; }
      }
      __syncthreads();
    }
    if (tid == 0) { topv[it] = rv[0]; topidx[it] = ri[0]; vals[ri[0]] = -1e30f; }
    __syncthreads();
  }
  const int c65 = cnt65;
  int vk = KK_ - 10; if (c65 > vk) vk = c65; if (vk > KK_) vk = KK_;
  if (tid < KK_) {
    const float val = (tid < vk) ? topv[tid] : 0.f;
    topi[(size_t)b * KK_ + tid] = topidx[tid];
    if (val > 0.f)
      sel[(size_t)b * KK_ * NS_ + (size_t)tid * NS_ + topidx[tid]] = 1.f;
  }
}

// ---------------- host orchestration ----------------
static void launch_gemm(const bf16_t* A, const bf16_t* Bw, int M, int N, int K,
                        const float* bias, int mode, const int* tt, int ttv,
                        float* outf, bf16_t* outbf, hipStream_t s) {
  gemm_bf16_kernel<<<dim3(N / 128, M / 128), 256, 0, s>>>(A, Bw, M, N, K, bias, mode, tt, ttv, outf, outbf);
}

extern "C" void kernel_launch(void* const* d_in, const int* in_sizes, int n_in,
                              void* d_out, int out_size, void* d_ws, size_t ws_size,
                              hipStream_t stream) {
  (void)in_sizes; (void)n_in; (void)out_size; (void)ws_size;
  const float* x_b   = (const float*)d_in[0];
  const float* x_s   = (const float*)d_in[1];
  const int*   ttyp  = (const int*)d_in[2];
  const int*   bidx  = (const int*)d_in[3];
  const float* ln1_w = (const float*)d_in[4];
  const float* ln1_b = (const float*)d_in[5];
  const float* ln2_w = (const float*)d_in[6];
  const float* ln2_b = (const float*)d_in[7];
  const float* ln3_w = (const float*)d_in[8];
  const float* ln3_b = (const float*)d_in[9];
  const float* Wq    = (const float*)d_in[10];
  const float* Wkv   = (const float*)d_in[11];
  const float* es_w1 = (const float*)d_in[12];
  const float* es_b1 = (const float*)d_in[13];
  const float* es_w2 = (const float*)d_in[14];
  const float* es_b2 = (const float*)d_in[15];
  const float* el_w1 = (const float*)d_in[16];
  const float* el_b1 = (const float*)d_in[17];
  const float* el_w2 = (const float*)d_in[18];
  const float* el_b2 = (const float*)d_in[19];
  const float* ml_w1 = (const float*)d_in[20];
  const float* ml_b1 = (const float*)d_in[21];
  const float* ml_w2 = (const float*)d_in[22];
  const float* ml_b2 = (const float*)d_in[23];

  // workspace layout
  char* ws = (char*)d_ws;
  size_t off = 0;
  auto alloc = [&](size_t bytes) -> void* {
    void* p = ws + off;
    off = (off + bytes + 255) & ~(size_t)255;
    return p;
  };
  bf16_t* wq_bf = (bf16_t*)alloc((size_t)C_ * C_ * 2);
  bf16_t* wkv_bf = (bf16_t*)alloc((size_t)C_ * 2 * C_ * 2);
  bf16_t* esw1_bf = (bf16_t*)alloc((size_t)C_ * HID_ * 2);
  bf16_t* esw2_bf = (bf16_t*)alloc((size_t)HID_ * C_ * 2);
  bf16_t* elw1_bf = (bf16_t*)alloc((size_t)C_ * HID_ * 2);
  bf16_t* elw2_bf = (bf16_t*)alloc((size_t)HID_ * C_ * 2);
  bf16_t* mlw1_bf = (bf16_t*)alloc((size_t)C_ * HID_ * 2);
  bf16_t* mlw2_bf = (bf16_t*)alloc((size_t)HID_ * C_ * 2);
  bf16_t* xs_n  = (bf16_t*)alloc((size_t)B_ * NS_ * C_ * 2);
  bf16_t* xb_n  = (bf16_t*)alloc((size_t)B_ * NB_ * C_ * 2);
  bf16_t* qbf   = (bf16_t*)alloc((size_t)B_ * NS_ * C_ * 2);
  bf16_t* kvbf  = (bf16_t*)alloc((size_t)B_ * NB_ * 2 * C_ * 2);
  bf16_t* ctxbf = (bf16_t*)alloc((size_t)B_ * NS_ * C_ * 2);
  bf16_t* h1bf  = (bf16_t*)alloc((size_t)B_ * NS_ * HID_ * 2);
  bf16_t* xn3   = (bf16_t*)alloc((size_t)B_ * NS_ * C_ * 2);
  float*  xbuf  = (float*)alloc((size_t)B_ * NS_ * C_ * 4);
  float*  invb  = (float*)alloc((size_t)B_ * NS_ * 4);
  float*  toks  = (float*)alloc((size_t)B_ * NHALF_ * C_ * 4);
  float*  aggb  = (float*)alloc((size_t)B_ * NHALF_ * C_ * 4);
  float*  kws   = (float*)alloc((size_t)B_ * C_ * 4);
  float*  hasws = (float*)alloc((size_t)B_ * 4);

  // output layout (tuple flattened)
  float* out_f = (float*)d_out;
  float* selo  = out_f;                                   // 8*128*1024
  int*   topio = (int*)(out_f + (size_t)B_ * KK_ * NS_);  // 8*128 (int32 bits)
  float* poso  = out_f + (size_t)B_ * KK_ * NS_ + (size_t)B_ * KK_;
  float* xout  = poso + (size_t)B_ * NS_;
  float* kout  = xout + (size_t)B_ * NS_ * C_;

  auto cvt = [&](const float* src, bf16_t* dst, long n) {
    f32_to_bf16_kernel<<<(unsigned)((n + 255) / 256), 256, 0, stream>>>(src, dst, n);
  };
  cvt(Wq, wq_bf, (long)C_ * C_);
  cvt(Wkv, wkv_bf, (long)C_ * 2 * C_);
  cvt(es_w1, esw1_bf, (long)C_ * HID_);
  cvt(es_w2, esw2_bf, (long)HID_ * C_);
  cvt(el_w1, elw1_bf, (long)C_ * HID_);
  cvt(el_w2, elw2_bf, (long)HID_ * C_);
  cvt(ml_w1, mlw1_bf, (long)C_ * HID_);
  cvt(ml_w2, mlw2_bf, (long)HID_ * C_);

  ln_cast_kernel<<<B_ * NS_, 256, 0, stream>>>(x_s, ln1_w, ln1_b, xs_n, C_);
  ln_cast_kernel<<<B_ * NB_, 256, 0, stream>>>(x_b, ln2_w, ln2_b, xb_n, C_);

  launch_gemm(xs_n, wq_bf, B_ * NS_, C_, C_, nullptr, 0, nullptr, 0, nullptr, qbf, stream);
  launch_gemm(xb_n, wkv_bf, B_ * NB_, 2 * C_, C_, nullptr, 0, nullptr, 0, nullptr, kvbf, stream);

  attention_kernel<<<dim3(NS_ / 128, NH_, B_), 256, 0, stream>>>(qbf, kvbf, ctxbf);

  copy_f32_kernel<<<(unsigned)(((long)B_ * NS_ * C_ + 255) / 256), 256, 0, stream>>>(
      x_s, xbuf, (long)B_ * NS_ * C_);

  // MoE: short + long branches, masked residual add into xbuf
  launch_gemm(ctxbf, esw1_bf, B_ * NS_, HID_, C_, es_b1, 1, nullptr, 0, nullptr, h1bf, stream);
  launch_gemm(h1bf, esw2_bf, B_ * NS_, C_, HID_, es_b2, 2, ttyp, 0, xbuf, nullptr, stream);
  launch_gemm(ctxbf, elw1_bf, B_ * NS_, HID_, C_, el_b1, 1, nullptr, 0, nullptr, h1bf, stream);
  launch_gemm(h1bf, elw2_bf, B_ * NS_, C_, HID_, el_b2, 2, ttyp, 1, xbuf, nullptr, stream);

  // MLP with residual
  ln_cast_kernel<<<B_ * NS_, 256, 0, stream>>>(xbuf, ln3_w, ln3_b, xn3, C_);
  launch_gemm(xn3, mlw1_bf, B_ * NS_, HID_, C_, ml_b1, 1, nullptr, 0, nullptr, h1bf, stream);
  launch_gemm(h1bf, mlw2_bf, B_ * NS_, C_, HID_, ml_b2, 3, nullptr, 0, xbuf, nullptr, stream);

  hipMemcpyAsync(xout, xbuf, (size_t)B_ * NS_ * C_ * 4, hipMemcpyDeviceToDevice, stream);

  // scoring head
  rownorm_inv_kernel<<<B_ * NS_, 256, 0, stream>>>(xbuf, invb, C_);
  gather_tokens_kernel<<<B_ * NHALF_, 256, 0, stream>>>(x_b, bidx, toks);
  token_attn_kernel<<<B_ * NHALF_, 256, 0, stream>>>(toks, bidx, aggb);
  kernel_reduce_kernel<<<B_, 256, 0, stream>>>(aggb, bidx, kout, kws, hasws);
  pos_kernel<<<B_ * NS_, 256, 0, stream>>>(xbuf, invb, kws, hasws, poso);

  zero_f32_kernel<<<(unsigned)(((long)B_ * KK_ * NS_ + 255) / 256), 256, 0, stream>>>(
      selo, (long)B_ * KK_ * NS_);
  topk_kernel<<<B_, 256, 0, stream>>>(poso, selo, topio);
}
